// MultiHeadAttention_38749194945230
// MI455X (gfx1250) — matrix operand order repaired and checked by reading, not compiled, not run on hardware
//
#include <hip/hip_runtime.h>
#include <cstdint>
#include <cstddef>

#define B_  2
#define S_  2048
#define D_  2048
#define H_  16
#define HD_ 128

// ---------------------------------------------------------------------------
// CDNA5 WMMA types (wave32): bf16 A/B fragments (16 halfs = 8 VGPRs),
// f32 C/D accumulators (8 floats = 8 VGPRs).
// ---------------------------------------------------------------------------
typedef __bf16        bf16x16 __attribute__((ext_vector_type(16)));
typedef float         f32x8   __attribute__((ext_vector_type(8)));
typedef float         f32x4   __attribute__((ext_vector_type(4)));
typedef unsigned int  u32x4   __attribute__((ext_vector_type(4)));

union BFrag {
    bf16x16      v;
    unsigned int w[8];
    u32x4        q[2];
};

__device__ __forceinline__ unsigned int f32_to_bf16_bits(float f) {
    union { float f; unsigned int u; } c; c.f = f;
    unsigned int u = c.u;
    u += 0x7fffu + ((u >> 16) & 1u);   // round-to-nearest-even
    return u >> 16;
}
__device__ __forceinline__ unsigned int pack2_bf16(float a, float b) {
    return f32_to_bf16_bits(a) | (f32_to_bf16_bits(b) << 16);
}
// Convert 8 contiguous f32 -> 8 bf16 packed in a u32x4.
__device__ __forceinline__ u32x4 cvt8_bf16v(const float* __restrict__ p) {
    f32x4 a = *(const f32x4*)p;
    f32x4 b = *(const f32x4*)(p + 4);
    u32x4 r;
    r.x = pack2_bf16(a.x, a.y);
    r.y = pack2_bf16(a.z, a.w);
    r.z = pack2_bf16(b.x, b.y);
    r.w = pack2_bf16(b.z, b.w);
    return r;
}

// ---------------------------------------------------------------------------
// One-time f32 -> bf16 tensor conversion (memory-bound, off the hot path).
// One thread per 8 elements.
// ---------------------------------------------------------------------------
__global__ __launch_bounds__(256) void f32_to_bf16_kernel(
    const float* __restrict__ src, unsigned short* __restrict__ dst, int n8)
{
    const int i = blockIdx.x * blockDim.x + threadIdx.x;
    if (i >= n8) return;
    *(u32x4*)(dst + (size_t)i * 8) = cvt8_bf16v(src + (size_t)i * 8);
}

// ---------------------------------------------------------------------------
// GEMM: Y[M,N] = X[M,K] @ W[N,K]^T + bias[N]   (bf16 in, f32 WMMA accum)
// Block: 256 threads = 8 waves; block tile 128x128; wave tile 64x32.
// Inner loop is pure b128 loads + v_wmma (no conversion VALU).
// A-frag, contiguous k as loaded here: lane(l16,g): row=l16, K = 16g + j.
// B-frag, contiguous k as loaded here (same as A): lane(l16,g): col=l16, K = 16g + j.
// C/D layout: lane(l16,g): n=l16, m = 8g + e.
// ---------------------------------------------------------------------------
__global__ __launch_bounds__(256) void gemm_bf16_kernel(
    const unsigned short* __restrict__ X, const unsigned short* __restrict__ W,
    const float* __restrict__ bias, float* __restrict__ Y,
    int M, int N, int K)
{
    const int lane = threadIdx.x & 31;
    const int wave = threadIdx.x >> 5;
    const int g    = lane >> 4;
    const int l16  = lane & 15;
    const int wm   = blockIdx.x * 128 + (wave >> 2) * 64;
    const int wn   = blockIdx.y * 128 + (wave & 3) * 32;

    f32x8 acc[4][2];
#pragma unroll
    for (int mt = 0; mt < 4; ++mt)
#pragma unroll
        for (int nt = 0; nt < 2; ++nt)
#pragma unroll
            for (int e = 0; e < 8; ++e) acc[mt][nt][e] = 0.f;

    const unsigned short* xrow[4];
#pragma unroll
    for (int mt = 0; mt < 4; ++mt)
        xrow[mt] = X + (size_t)(wm + mt * 16 + l16) * K;
    const unsigned short* wrow[2];
#pragma unroll
    for (int nt = 0; nt < 2; ++nt)
        wrow[nt] = W + (size_t)(wn + nt * 16 + l16) * K;

#pragma unroll 2
    for (int k0 = 0; k0 < K; k0 += 32) {
        BFrag bf[2];
#pragma unroll
        for (int nt = 0; nt < 2; ++nt) {
            const unsigned short* p = wrow[nt] + k0 + 16 * g;
            bf[nt].q[0] = *(const u32x4*)(p);
            bf[nt].q[1] = *(const u32x4*)(p + 8);
        }
#pragma unroll
        for (int mt = 0; mt < 4; ++mt) {
            BFrag af;
            const unsigned short* p = xrow[mt] + k0 + 16 * g;
            af.q[0] = *(const u32x4*)(p);
            af.q[1] = *(const u32x4*)(p + 8);
#pragma unroll
            for (int nt = 0; nt < 2; ++nt)
                acc[mt][nt] = __builtin_amdgcn_wmma_f32_16x16x32_bf16(
                    false, af.v, false, bf[nt].v, (short)0, acc[mt][nt], false, false);
        }
    }

#pragma unroll
    for (int nt = 0; nt < 2; ++nt) {
        const int n = wn + nt * 16 + l16;
        const float bv = bias[n];
#pragma unroll
        for (int mt = 0; mt < 4; ++mt) {
            const int mbase = wm + mt * 16 + 8 * g;
#pragma unroll
            for (int e = 0; e < 8; ++e)
                Y[(size_t)(mbase + e) * N + n] = acc[mt][nt][e] + bv;
        }
    }
}

// ---------------------------------------------------------------------------
// RoPE + layout/precision conversion.
// Writes bf16 Q,K in [b,h,s,hd]; bf16 V^T in [b,h,hd,s];
// f32 roped-K and f32 V (both [b,h,s,hd]) straight into d_out.
// One thread per (b,s,h,pair).
// ---------------------------------------------------------------------------
__global__ __launch_bounds__(256) void rope_pack_kernel(
    const float* __restrict__ Qf, const float* __restrict__ Kf,
    const float* __restrict__ Vf,
    const float* __restrict__ Ct, const float* __restrict__ St,
    unsigned short* __restrict__ Qb, unsigned short* __restrict__ Kb,
    unsigned short* __restrict__ Vt,
    float* __restrict__ Kout, float* __restrict__ Vout)
{
    const int P = HD_ / 2;
    const int idx = blockIdx.x * blockDim.x + threadIdx.x;
    const int p = idx % P;
    const int h = (idx / P) % H_;
    const int s = (idx / (P * H_)) % S_;
    const int b = idx / (P * H_ * S_);

    const size_t src = ((size_t)(b * S_ + s) * H_ + h) * HD_ + 2 * p;   // [b,s,h,hd]
    const size_t dst = (((size_t)b * H_ + h) * S_ + s) * HD_ + 2 * p;   // [b,h,s,hd]

    const float c  = Ct[s * P + p];
    const float sn = St[s * P + p];

    const float qr = Qf[src], qi = Qf[src + 1];
    const float kr = Kf[src], ki = Kf[src + 1];
    const float q0 = qr * c - qi * sn, q1 = qr * sn + qi * c;
    const float k0 = kr * c - ki * sn, k1 = kr * sn + ki * c;

    Qb[dst]     = (unsigned short)f32_to_bf16_bits(q0);
    Qb[dst + 1] = (unsigned short)f32_to_bf16_bits(q1);
    Kb[dst]     = (unsigned short)f32_to_bf16_bits(k0);
    Kb[dst + 1] = (unsigned short)f32_to_bf16_bits(k1);
    Kout[dst]     = k0;
    Kout[dst + 1] = k1;

    const float v0 = Vf[src], v1 = Vf[src + 1];
    Vout[dst]     = v0;
    Vout[dst + 1] = v1;

    const size_t vtb = (((size_t)b * H_ + h) * HD_) * S_;               // [b,h,hd,s]
    Vt[vtb + (size_t)(2 * p)     * S_ + s] = (unsigned short)f32_to_bf16_bits(v0);
    Vt[vtb + (size_t)(2 * p + 1) * S_ + s] = (unsigned short)f32_to_bf16_bits(v1);
}

// ---------------------------------------------------------------------------
// Causal flash attention, register-resident per wave.
// Grid: (S/128, B*H). Block 256 = 8 waves; each wave owns 16 queries.
// Computes S^T = K @ Q^T (softmax reduction = 16 in-lane + shfl_xor(16)),
// builds P^T B-fragment via one cross-half shuffle, accumulates
// O^T = V^T @ P^T with per-lane (= per-query) online softmax rescaling.
// Emits bf16 output directly for the W_o projection.
// ---------------------------------------------------------------------------
__global__ __launch_bounds__(256) void flash_attn_kernel(
    const unsigned short* __restrict__ Qb, const unsigned short* __restrict__ Kb,
    const unsigned short* __restrict__ Vt, unsigned short* __restrict__ AOb)
{
    const int lane = threadIdx.x & 31;
    const int wave = threadIdx.x >> 5;
    const int g    = lane >> 4;
    const int l16  = lane & 15;
    const int qrow0 = blockIdx.x * 128 + wave * 16;
    const int bh = blockIdx.y;
    const int b = bh >> 4;             // H_ == 16
    const int h = bh & 15;
    const int myq = qrow0 + l16;

    // Q^T B-fragments for the whole HD=128: 4 chunks of K=32.
    const unsigned short* qp = Qb + (((size_t)b * H_ + h) * S_ + myq) * HD_;
    BFrag qf[4];
#pragma unroll
    for (int kc = 0; kc < 4; ++kc) {
        qf[kc].q[0] = *(const u32x4*)(qp + kc * 32 + 16 * g);
        qf[kc].q[1] = *(const u32x4*)(qp + kc * 32 + 16 * g + 8);
    }

    const unsigned short* kbase = Kb + (((size_t)b * H_ + h) * S_) * HD_;
    const unsigned short* vbase = Vt + (((size_t)b * H_ + h) * HD_) * S_;

    f32x8 o[8];
#pragma unroll
    for (int dt = 0; dt < 8; ++dt)
#pragma unroll
        for (int e = 0; e < 8; ++e) o[dt][e] = 0.f;

    float mrun = -__builtin_inff();
    float lrun = 0.f;
    const float scale = 0.088388347648318447f;   // 1/sqrt(HD)

    for (int kb0 = 0; kb0 <= qrow0 + 15; kb0 += 32) {
        // ---- scores: S^T tiles (keys kb0+16*mt .. +15, 16 queries) ----
        f32x8 sc[2];
#pragma unroll
        for (int mt = 0; mt < 2; ++mt) {
            f32x8 c;
#pragma unroll
            for (int e = 0; e < 8; ++e) c[e] = 0.f;
            const unsigned short* kp = kbase + (size_t)(kb0 + mt * 16 + l16) * HD_;
#pragma unroll
            for (int kc = 0; kc < 4; ++kc) {
                BFrag a;
                a.q[0] = *(const u32x4*)(kp + kc * 32 + 16 * g);
                a.q[1] = *(const u32x4*)(kp + kc * 32 + 16 * g + 8);
                c = __builtin_amdgcn_wmma_f32_16x16x32_bf16(
                        false, a.v, false, qf[kc].v, (short)0, c, false, false);
            }
            sc[mt] = c;
        }

        // ---- scale + causal mask + online softmax ----
        float tmax = -__builtin_inff();
        float pv[2][8];
#pragma unroll
        for (int mt = 0; mt < 2; ++mt)
#pragma unroll
            for (int e = 0; e < 8; ++e) {
                const int key = kb0 + mt * 16 + 8 * g + e;
                float s = sc[mt][e] * scale;
                s = (key > myq) ? -__builtin_inff() : s;
                pv[mt][e] = s;
                tmax = fmaxf(tmax, s);
            }
        tmax = fmaxf(tmax, __shfl_xor(tmax, 16, 32));
        const float mnew = fmaxf(mrun, tmax);
        const float corr = __expf(mrun - mnew);
        float tsum = 0.f;
#pragma unroll
        for (int mt = 0; mt < 2; ++mt)
#pragma unroll
            for (int e = 0; e < 8; ++e) {
                const float pe = __expf(pv[mt][e] - mnew);
                pv[mt][e] = pe;
                tsum += pe;
            }
        tsum += __shfl_xor(tsum, 16, 32);
        lrun = lrun * corr + tsum;
        mrun = mnew;

#pragma unroll
        for (int dt = 0; dt < 8; ++dt)
#pragma unroll
            for (int e = 0; e < 8; ++e) o[dt][e] *= corr;

        // ---- P^T C-layout -> B-fragment layout via one xor-16 exchange ----
        unsigned int pk0[4], pk1[4];
#pragma unroll
        for (int i = 0; i < 4; ++i) {
            pk0[i] = pack2_bf16(pv[0][2 * i], pv[0][2 * i + 1]);
            pk1[i] = pack2_bf16(pv[1][2 * i], pv[1][2 * i + 1]);
        }
        BFrag pf;
#pragma unroll
        for (int i = 0; i < 4; ++i) {
            const unsigned int snd = g ? pk0[i] : pk1[i];
            const unsigned int rcv = __shfl_xor(snd, 16, 32);
            pf.w[i]     = g ? rcv    : pk0[i];
            pf.w[4 + i] = g ? pk1[i] : rcv;
        }

        // ---- O^T += V^T @ P^T ----
#pragma unroll
        for (int dt = 0; dt < 8; ++dt) {
            const unsigned short* vp = vbase + (size_t)(dt * 16 + l16) * S_ + kb0;
            BFrag a;
            a.q[0] = *(const u32x4*)(vp + 16 * g);
            a.q[1] = *(const u32x4*)(vp + 16 * g + 8);
            o[dt] = __builtin_amdgcn_wmma_f32_16x16x32_bf16(
                        false, a.v, false, pf.v, (short)0, o[dt], false, false);
        }
    }

    // ---- finalize: lane n == query, element m == head dim; emit bf16 ----
    const float inv = 1.f / lrun;
    unsigned short* aop = AOb + (((size_t)b * S_ + myq) * H_ + h) * HD_;  // [b,s,d]
#pragma unroll
    for (int dt = 0; dt < 8; ++dt) {
        u32x4 r;
        r.x = pack2_bf16(o[dt][0] * inv, o[dt][1] * inv);
        r.y = pack2_bf16(o[dt][2] * inv, o[dt][3] * inv);
        r.z = pack2_bf16(o[dt][4] * inv, o[dt][5] * inv);
        r.w = pack2_bf16(o[dt][6] * inv, o[dt][7] * inv);
        *(u32x4*)(aop + dt * 16 + 8 * g) = r;
    }
}

// ---------------------------------------------------------------------------
// Host-side orchestration (graph-capture safe: launches only).
// ---------------------------------------------------------------------------
extern "C" void kernel_launch(void* const* d_in, const int* in_sizes, int n_in,
                              void* d_out, int out_size, void* d_ws, size_t ws_size,
                              hipStream_t stream)
{
    (void)in_sizes; (void)n_in; (void)out_size; (void)ws_size;

    const float* x   = (const float*)d_in[0];
    const float* wq  = (const float*)d_in[1];
    const float* wqb = (const float*)d_in[2];
    const float* wk  = (const float*)d_in[3];
    const float* wkb = (const float*)d_in[4];
    const float* wv  = (const float*)d_in[5];
    const float* wvb = (const float*)d_in[6];
    const float* wo  = (const float*)d_in[7];
    const float* wob = (const float*)d_in[8];
    const float* fc  = (const float*)d_in[9];
    const float* fs  = (const float*)d_in[10];

    float* out = (float*)d_out;
    const size_t NE = (size_t)B_ * S_ * D_;   // 8.39M elements (x / q / k / v)
    const size_t DD = (size_t)D_ * D_;        // 4.19M elements (weights)
    float* kout = out + NE;                   // d_out: [out | k | v]
    float* vout = kout + NE;

    char* w = (char*)d_ws;
    float* qf = (float*)w;                      w += NE * sizeof(float);
    float* kf = (float*)w;                      w += NE * sizeof(float);
    float* vf = (float*)w;                      w += NE * sizeof(float);
    unsigned short* qb  = (unsigned short*)w;   w += NE * 2;
    unsigned short* kb  = (unsigned short*)w;   w += NE * 2;
    unsigned short* vt  = (unsigned short*)w;   w += NE * 2;
    unsigned short* xb  = (unsigned short*)w;   w += NE * 2;
    unsigned short* wqh = (unsigned short*)w;   w += DD * 2;
    unsigned short* wkh = (unsigned short*)w;   w += DD * 2;
    unsigned short* wvh = (unsigned short*)w;   w += DD * 2;
    unsigned short* woh = (unsigned short*)w;   w += DD * 2;
    unsigned short* aob = (unsigned short*)qf;  // alias: qf dead after rope

    dim3 blk(256);

    // One-time bf16 conversions (memory-bound prologue).
    f32_to_bf16_kernel<<<(int)(NE / 8 / 256), blk, 0, stream>>>(x,  xb,  (int)(NE / 8));
    f32_to_bf16_kernel<<<(int)(DD / 8 / 256), blk, 0, stream>>>(wq, wqh, (int)(DD / 8));
    f32_to_bf16_kernel<<<(int)(DD / 8 / 256), blk, 0, stream>>>(wk, wkh, (int)(DD / 8));
    f32_to_bf16_kernel<<<(int)(DD / 8 / 256), blk, 0, stream>>>(wv, wvh, (int)(DD / 8));
    f32_to_bf16_kernel<<<(int)(DD / 8 / 256), blk, 0, stream>>>(wo, woh, (int)(DD / 8));

    const int M = B_ * S_;                    // 4096
    dim3 ggrid(M / 128, D_ / 128);            // (32, 16)

    gemm_bf16_kernel<<<ggrid, blk, 0, stream>>>(xb, wqh, wqb, qf, M, D_, D_);
    gemm_bf16_kernel<<<ggrid, blk, 0, stream>>>(xb, wkh, wkb, kf, M, D_, D_);
    gemm_bf16_kernel<<<ggrid, blk, 0, stream>>>(xb, wvh, wvb, vf, M, D_, D_);

    const int nrope = B_ * S_ * H_ * (HD_ / 2);
    rope_pack_kernel<<<nrope / 256, blk, 0, stream>>>(qf, kf, vf, fc, fs,
                                                      qb, kb, vt, kout, vout);

    flash_attn_kernel<<<dim3(S_ / 128, B_ * H_), blk, 0, stream>>>(qb, kb, vt, aob);

    gemm_bf16_kernel<<<ggrid, blk, 0, stream>>>(aob, woh, wob, out, M, D_, D_);
}
